// MyActor_27127013441772
// MI455X (gfx1250) — matrix operand order repaired and checked by reading, not compiled, not run on hardware
//
#include <hip/hip_runtime.h>

typedef __attribute__((ext_vector_type(16))) __bf16 v16bf;
typedef __attribute__((ext_vector_type(8)))  float  v8f;
typedef __attribute__((ext_vector_type(4)))  unsigned short us4;
typedef __attribute__((ext_vector_type(4)))  int i4;

#define LN_EPS   1e-5f
#define NORM_EPS 1e-8f

#if defined(__HIP_DEVICE_COMPILE__) && defined(__has_builtin)
#  if __has_builtin(__builtin_amdgcn_global_load_async_to_lds_b128)
#    define HAVE_ASYNC_LDS 1
#  else
#    define HAVE_ASYNC_LDS 0
#  endif
#  if __has_builtin(__builtin_amdgcn_s_wait_asynccnt)
#    define HAVE_WAIT_ASYNC 1
#  else
#    define HAVE_WAIT_ASYNC 0
#  endif
#else
#  define HAVE_ASYNC_LDS 0
#  define HAVE_WAIT_ASYNC 0
#endif

// Global fragment-id bases. Each fragment = 512 halfs (32 lanes x 16 bf16 = 1KB).
// Fragment id within a weight = kc * NT + nt  (kc = K/32 chunk, nt = N/16 tile).
#define FB_W1S 0     // sur  W1:  64x128  KC=2 NT=8   -> 16 frags
#define FB_W2S 16    // sur  W2: 128x128  KC=4 NT=8   -> 32
#define FB_WP1 48    // proj W1: 192x128  KC=6 NT=8   -> 48
#define FB_WP2 96    // proj W2: 128x128  KC=4 NT=8   -> 32
#define FB_WM1 128   // modl W1: 128x256  KC=4 NT=16  -> 64
#define FB_WM2 192   // modl W2: 256x256  KC=8 NT=16  -> 128
#define FRAGS_TOTAL 320

union Frag {
  v16bf v;
  unsigned short us[16];
  uint4 q[2];
};

__device__ __forceinline__ unsigned short f2bf(float x){
  union { float f; unsigned u; } c; c.f = x;
  unsigned u = c.u;
  u += 0x7fffu + ((u >> 16) & 1u);          // round-to-nearest-even
  return (unsigned short)(u >> 16);
}
__device__ __forceinline__ float bf2f(unsigned short h){
  union { unsigned u; float f; } c; c.u = ((unsigned)h) << 16;
  return c.f;
}

__device__ __forceinline__ v8f v8f_zero(){
  v8f z = {0.f,0.f,0.f,0.f,0.f,0.f,0.f,0.f};
  return z;
}

__device__ __forceinline__ v8f wmma_bf16(v16bf a, v16bf b, v8f c){
  // D = A(16x32 bf16) * B(32x16 bf16) + C(16x16 f32)
  return __builtin_amdgcn_wmma_f32_16x16x32_bf16(false, a, false, b, (short)0, c, false, false);
}

__device__ __forceinline__ void wait_async0(){
#if HAVE_WAIT_ASYNC
  __builtin_amdgcn_s_wait_asynccnt(0);
#else
  asm volatile("s_wait_asynccnt 0" ::: "memory");
#endif
}

// B fragment: pre-swizzled in workspace; lane's 16 halfs are contiguous (32B).
__device__ __forceinline__ v16bf load_bfrag(const unsigned short* __restrict__ wsw, int fragId, int lane){
  const uint4* p = (const uint4*)(wsw + ((size_t)fragId << 9) + (lane << 4));
  Frag f; f.q[0] = p[0]; f.q[1] = p[1];
  return f.v;
}

// A fragment gathered from LDS (row-major bf16, 'stride' halfs per row).
// ISA 16-bit A 16x32 layout: lanes 0-15 M=lane, K pairs {0..7,16..23}; lanes 16-31 add +8.
__device__ __forceinline__ v16bf gather_afrag(const unsigned short* __restrict__ lds,
                                              int stride, int mBase, int kb, int lane){
  const int m    = mBase + (lane & 15);
  const int koff = kb + ((lane >> 4) << 3);
  Frag f;
  #pragma unroll
  for (int p = 0; p < 8; ++p){
    const int k = koff + ((p < 4) ? (p << 1) : (16 + ((p - 4) << 1)));
    const unsigned short* s = &lds[m * stride + k];
    f.us[2*p]     = s[0];
    f.us[2*p + 1] = s[1];
  }
  return f.v;
}

// Fused epilogue: bias + ReLU (+ optional LayerNorm over NT*16 columns) then bf16 store to LDS.
// C/D layout: lane L holds N = 16*t + (L&15); VGPR r holds row mBase + r + 8*(L>>4).
template<int NT, bool LN>
__device__ __forceinline__ void epilogue_store(v8f* acc,
    const float* __restrict__ bias, const float* __restrict__ gam, const float* __restrict__ bet,
    unsigned short* __restrict__ dst, int stride, int mBase, int lane)
{
  const int nl = lane & 15, half = lane >> 4;
  float bv[NT], gv[NT], bbv[NT];
  #pragma unroll
  for (int t = 0; t < NT; ++t){
    const int n = (t << 4) + nl;
    bv[t] = bias[n];
    if (LN){ gv[t] = gam[n]; bbv[t] = bet[n]; }
  }
  const float invN = 1.0f / (float)(NT * 16);
  #pragma unroll
  for (int r = 0; r < 8; ++r){
    float v[NT], s = 0.f, s2 = 0.f;
    #pragma unroll
    for (int t = 0; t < NT; ++t){
      float x = acc[t][r] + bv[t];
      x = x > 0.f ? x : 0.f;
      v[t] = x; s += x; s2 += x * x;
    }
    if (LN){
      #pragma unroll
      for (int mm = 1; mm < 16; mm <<= 1){ s += __shfl_xor(s, mm); s2 += __shfl_xor(s2, mm); }
      const float mu   = s * invN;
      const float rstd = rsqrtf(s2 * invN - mu * mu + LN_EPS);
      #pragma unroll
      for (int t = 0; t < NT; ++t) v[t] = (v[t] - mu) * rstd * gv[t] + bbv[t];
    }
    const int m = mBase + r + (half << 3);
    #pragma unroll
    for (int t = 0; t < NT; ++t) dst[m * stride + (t << 4) + nl] = f2bf(v[t]);
  }
}

// Epilogue for sur GEMM2: bias+ReLU+LN then masked row-pool into sPool[128].
__device__ __forceinline__ void epilogue_pool8(v8f* acc,
    const float* __restrict__ bias, const float* __restrict__ gam, const float* __restrict__ bet,
    float* __restrict__ sPool, int mBase, int lane, int nValid)
{
  const int nl = lane & 15, half = lane >> 4;
  float bv[8], gv[8], bbv[8], pool[8];
  #pragma unroll
  for (int t = 0; t < 8; ++t){
    const int n = (t << 4) + nl;
    bv[t] = bias[n]; gv[t] = gam[n]; bbv[t] = bet[n]; pool[t] = 0.f;
  }
  #pragma unroll
  for (int r = 0; r < 8; ++r){
    float v[8], s = 0.f, s2 = 0.f;
    #pragma unroll
    for (int t = 0; t < 8; ++t){
      float x = acc[t][r] + bv[t]; x = x > 0.f ? x : 0.f;
      v[t] = x; s += x; s2 += x * x;
    }
    #pragma unroll
    for (int mm = 1; mm < 16; mm <<= 1){ s += __shfl_xor(s, mm); s2 += __shfl_xor(s2, mm); }
    const float mu   = s * 0.0078125f;                        // /128
    const float rstd = rsqrtf(s2 * 0.0078125f - mu * mu + LN_EPS);
    const int m = mBase + r + (half << 3);
    if (m < nValid){
      #pragma unroll
      for (int t = 0; t < 8; ++t) pool[t] += (v[t] - mu) * rstd * gv[t] + bbv[t];
    }
  }
  #pragma unroll
  for (int t = 0; t < 8; ++t) pool[t] += __shfl_xor(pool[t], 16);
  if (lane < 16){
    #pragma unroll
    for (int t = 0; t < 8; ++t) atomicAdd(&sPool[(t << 4) + lane], pool[t]);
  }
}

// ---------------- prep: f32 weights -> bf16 WMMA B-fragment layout ----------------
__global__ __launch_bounds__(32) void prep_kernel(
    const float* __restrict__ W1s, const float* __restrict__ W2s,
    const float* __restrict__ Wp1, const float* __restrict__ Wp2,
    const float* __restrict__ Wm1, const float* __restrict__ Wm2,
    unsigned short* __restrict__ wsw)
{
  const int f = blockIdx.x, lane = threadIdx.x;
  const float* src; int N, NT, local;
  if      (f < 16)  { src = W1s; N = 128; NT = 8;  local = f;       }
  else if (f < 48)  { src = W2s; N = 128; NT = 8;  local = f - 16;  }
  else if (f < 96)  { src = Wp1; N = 128; NT = 8;  local = f - 48;  }
  else if (f < 128) { src = Wp2; N = 128; NT = 8;  local = f - 96;  }
  else if (f < 192) { src = Wm1; N = 256; NT = 16; local = f - 128; }
  else              { src = Wm2; N = 256; NT = 16; local = f - 192; }
  const int kc = local / NT, nt = local % NT;
  const int n  = (nt << 4) + (lane & 15);
  const int kb = (kc << 5) + ((lane >> 4) << 3);   // B layout: lane half selects K+8; elems 8..15 at K+16
  unsigned short* dst = wsw + ((size_t)f << 9) + (lane << 4);
  #pragma unroll
  for (int e = 0; e < 16; ++e)
    dst[e] = f2bf(src[(size_t)(kb + e + ((e >> 3) << 3)) * N + n]);
}

// ---------------- sur branch: per-sample 128x64 -> MLP -> masked mean ----------------
__global__ __launch_bounds__(256) void sur_kernel(
    const float* __restrict__ sur_data, const int* __restrict__ sur_n,
    const float* __restrict__ mean, const float* __restrict__ stdv,
    const float* __restrict__ b1, const float* __restrict__ g1, const float* __restrict__ bb1,
    const float* __restrict__ b2, const float* __restrict__ g2, const float* __restrict__ bb2,
    const unsigned short* __restrict__ wsw, float* __restrict__ sur_feat)
{
  __shared__ __align__(16) unsigned short sX[128 * 72];    // normalized input, bf16, padded
  __shared__ __align__(16) unsigned short sH[128 * 130];   // raw f32 stage, then hidden bf16
  __shared__ float sPool[128];
  __shared__ float sPar[6 * 128];            // b1,g1,bb1,b2,g2,bb2
  const int b = blockIdx.x, tid = threadIdx.x;

  for (int i = tid; i < 768; i += 256){
    const int which = i >> 7, j = i & 127;
    const float* p = (which == 0) ? b1 : (which == 1) ? g1 : (which == 2) ? bb1
                   : (which == 3) ? b2 : (which == 4) ? g2 : bb2;
    sPar[i] = p[j];
  }
  if (tid < 128) sPool[tid] = 0.f;

  // stage sur_data[b] (128x64 f32) and normalize -> sX bf16
  const float4* src4 = (const float4*)(sur_data + (size_t)b * 8192);
#if HAVE_ASYNC_LDS
  {
    // async-copy raw f32 tile into the (not-yet-used) sH region, 16B per lane per op
    float* raw = (float*)sH;
    #pragma unroll
    for (int it = 0; it < 8; ++it){
      const int idx = tid + (it << 8);                  // float4 index, 2048 total
      __builtin_amdgcn_global_load_async_to_lds_b128(
          (__attribute__((address_space(1))) i4*)(src4 + idx),
          (__attribute__((address_space(3))) i4*)(raw + idx * 4),
          0, 0);
    }
    wait_async0();
    __syncthreads();
    const float4* raw4 = (const float4*)raw;
    for (int idx = tid; idx < 2048; idx += 256){
      const int r = idx >> 4, k4 = (idx & 15) << 2;     // row, starting feature
      const float4 x = raw4[idx];
      us4 o;
      o.x = f2bf((x.x - mean[k4+0]) / (stdv[k4+0] + NORM_EPS));
      o.y = f2bf((x.y - mean[k4+1]) / (stdv[k4+1] + NORM_EPS));
      o.z = f2bf((x.z - mean[k4+2]) / (stdv[k4+2] + NORM_EPS));
      o.w = f2bf((x.w - mean[k4+3]) / (stdv[k4+3] + NORM_EPS));
      *(us4*)&sX[r * 72 + k4] = o;
    }
  }
#else
  for (int idx = tid; idx < 2048; idx += 256){
    const int r = idx >> 4, k4 = (idx & 15) << 2;
    const float4 x = src4[idx];
    us4 o;
    o.x = f2bf((x.x - mean[k4+0]) / (stdv[k4+0] + NORM_EPS));
    o.y = f2bf((x.y - mean[k4+1]) / (stdv[k4+1] + NORM_EPS));
    o.z = f2bf((x.z - mean[k4+2]) / (stdv[k4+2] + NORM_EPS));
    o.w = f2bf((x.w - mean[k4+3]) / (stdv[k4+3] + NORM_EPS));
    *(us4*)&sX[r * 72 + k4] = o;
  }
#endif
  __syncthreads();

  const int wave = tid >> 5, lane = tid & 31, mBase = wave << 4;

  // GEMM1: [128,64] @ [64,128]
  v8f acc[8];
  #pragma unroll
  for (int t = 0; t < 8; ++t) acc[t] = v8f_zero();
  const v16bf a0 = gather_afrag(sX, 72, mBase, 0,  lane);
  const v16bf a1 = gather_afrag(sX, 72, mBase, 32, lane);
  #pragma unroll
  for (int t = 0; t < 8; ++t){
    acc[t] = wmma_bf16(a0, load_bfrag(wsw, FB_W1S + t,     lane), acc[t]);
    acc[t] = wmma_bf16(a1, load_bfrag(wsw, FB_W1S + 8 + t, lane), acc[t]);
  }
  __syncthreads();   // everyone done reading raw region before epilogue overwrites sH
  epilogue_store<8, true>(acc, sPar, sPar + 128, sPar + 256, sH, 130, mBase, lane);
  __syncthreads();

  // GEMM2: [128,128] @ [128,128]
  #pragma unroll
  for (int t = 0; t < 8; ++t) acc[t] = v8f_zero();
  v16bf af[4];
  #pragma unroll
  for (int c = 0; c < 4; ++c) af[c] = gather_afrag(sH, 130, mBase, c << 5, lane);
  #pragma unroll
  for (int t = 0; t < 8; ++t){
    #pragma unroll
    for (int c = 0; c < 4; ++c)
      acc[t] = wmma_bf16(af[c], load_bfrag(wsw, FB_W2S + (c << 3) + t, lane), acc[t]);
  }
  const int nValid = sur_n[b];
  epilogue_pool8(acc, sPar + 384, sPar + 512, sPar + 640, sPool, mBase, lane, nValid);
  __syncthreads();

  if (tid < 128) sur_feat[(size_t)b * 128 + tid] = sPool[tid] / (float)nValid;
}

// ---------------- ego branch: tiny, VALU ----------------
__global__ __launch_bounds__(64) void ego_kernel(
    const float* __restrict__ ego_data, const int* __restrict__ ego_n,
    const float* __restrict__ mean, const float* __restrict__ stdv,
    const float* __restrict__ W1, const float* __restrict__ c1,
    const float* __restrict__ g1, const float* __restrict__ bb1,
    const float* __restrict__ W2, const float* __restrict__ c2,
    const float* __restrict__ g2, const float* __restrict__ bb2,
    float* __restrict__ ego_feat)
{
  const int b = blockIdx.x, j = threadIdx.x;   // 64 threads
  __shared__ float sx[8][33];
  __shared__ float h1[8][65];
  __shared__ float h2[8][65];
  __shared__ float st[8][2];
  for (int i = j; i < 256; i += 64){
    const int r = i >> 5, k = i & 31;
    sx[r][k] = (ego_data[(size_t)b * 256 + i] - mean[k]) / (stdv[k] + NORM_EPS);
  }
  __syncthreads();
  for (int r = 0; r < 8; ++r){
    float a = c1[j];
    for (int k = 0; k < 32; ++k) a += sx[r][k] * W1[k * 64 + j];
    h1[r][j] = fmaxf(a, 0.f);
  }
  __syncthreads();
  if (j < 8){
    float s = 0.f, s2 = 0.f;
    for (int k = 0; k < 64; ++k){ const float x = h1[j][k]; s += x; s2 += x * x; }
    const float mu = s * 0.015625f;
    st[j][0] = mu; st[j][1] = rsqrtf(s2 * 0.015625f - mu * mu + LN_EPS);
  }
  __syncthreads();
  for (int r = 0; r < 8; ++r) h1[r][j] = (h1[r][j] - st[r][0]) * st[r][1] * g1[j] + bb1[j];
  __syncthreads();
  for (int r = 0; r < 8; ++r){
    float a = c2[j];
    for (int k = 0; k < 64; ++k) a += h1[r][k] * W2[k * 64 + j];
    h2[r][j] = fmaxf(a, 0.f);
  }
  __syncthreads();
  if (j < 8){
    float s = 0.f, s2 = 0.f;
    for (int k = 0; k < 64; ++k){ const float x = h2[j][k]; s += x; s2 += x * x; }
    const float mu = s * 0.015625f;
    st[j][0] = mu; st[j][1] = rsqrtf(s2 * 0.015625f - mu * mu + LN_EPS);
  }
  __syncthreads();
  const int n = ego_n[b];
  float s = 0.f;
  for (int r = 0; r < 8; ++r){
    const float v = (h2[r][j] - st[r][0]) * st[r][1] * g2[j] + bb2[j];
    if (r < n) s += v;
  }
  ego_feat[(size_t)b * 64 + j] = s / (float)n;
}

// ---------------- head: proj MLP + model, 32 samples/block, 16/wave ----------------
__global__ __launch_bounds__(64) void head_kernel(
    const float* __restrict__ surF, const float* __restrict__ egoF,
    const unsigned short* __restrict__ wsw,
    const float* __restrict__ pb1, const float* __restrict__ pg1, const float* __restrict__ pbb1,
    const float* __restrict__ pb2, const float* __restrict__ pg2, const float* __restrict__ pbb2,
    const float* __restrict__ mb1, const float* __restrict__ mb2,
    const float* __restrict__ W3, const float* __restrict__ b3,
    float* __restrict__ out)
{
  __shared__ unsigned short sAbuf[2][16 * 258];
  __shared__ unsigned short sBbuf[2][16 * 258];
  const int wave = threadIdx.x >> 5, lane = threadIdx.x & 31;
  const int sb = blockIdx.x * 32 + (wave << 4);
  unsigned short* A  = sAbuf[wave];
  unsigned short* Bu = sBbuf[wave];

  // stage concat(sur_feat, ego_feat) as bf16: [16,192]
  for (int i = lane; i < 16 * 192; i += 32){
    const int m = i / 192, c = i % 192;
    const float v = (c < 128) ? surF[(size_t)(sb + m) * 128 + c]
                              : egoF[(size_t)(sb + m) * 64 + (c - 128)];
    A[m * 258 + c] = f2bf(v);
  }
  __syncthreads();

  // proj layer 1: K=192 -> N=128, ReLU+LN
  {
    v8f acc[8];
    #pragma unroll
    for (int t = 0; t < 8; ++t) acc[t] = v8f_zero();
    v16bf af[6];
    #pragma unroll
    for (int c = 0; c < 6; ++c) af[c] = gather_afrag(A, 258, 0, c << 5, lane);
    #pragma unroll
    for (int t = 0; t < 8; ++t){
      #pragma unroll
      for (int c = 0; c < 6; ++c)
        acc[t] = wmma_bf16(af[c], load_bfrag(wsw, FB_WP1 + (c << 3) + t, lane), acc[t]);
    }
    epilogue_store<8, true>(acc, pb1, pg1, pbb1, Bu, 258, 0, lane);
  }
  __syncthreads();

  // proj layer 2: K=128 -> N=128, ReLU+LN
  {
    v8f acc[8];
    #pragma unroll
    for (int t = 0; t < 8; ++t) acc[t] = v8f_zero();
    v16bf af[4];
    #pragma unroll
    for (int c = 0; c < 4; ++c) af[c] = gather_afrag(Bu, 258, 0, c << 5, lane);
    #pragma unroll
    for (int t = 0; t < 8; ++t){
      #pragma unroll
      for (int c = 0; c < 4; ++c)
        acc[t] = wmma_bf16(af[c], load_bfrag(wsw, FB_WP2 + (c << 3) + t, lane), acc[t]);
    }
    epilogue_store<8, true>(acc, pb2, pg2, pbb2, A, 258, 0, lane);
  }
  __syncthreads();

  // model layer 1: K=128 -> N=256, ReLU
  {
    v8f acc[16];
    #pragma unroll
    for (int t = 0; t < 16; ++t) acc[t] = v8f_zero();
    v16bf af[4];
    #pragma unroll
    for (int c = 0; c < 4; ++c) af[c] = gather_afrag(A, 258, 0, c << 5, lane);
    #pragma unroll
    for (int t = 0; t < 16; ++t){
      #pragma unroll
      for (int c = 0; c < 4; ++c)
        acc[t] = wmma_bf16(af[c], load_bfrag(wsw, FB_WM1 + (c << 4) + t, lane), acc[t]);
    }
    epilogue_store<16, false>(acc, mb1, mb1, mb1, Bu, 258, 0, lane);
  }
  __syncthreads();

  // model layer 2: K=256 -> N=256, ReLU (gather A inside loop to bound VGPRs)
  {
    v8f acc[16];
    #pragma unroll
    for (int t = 0; t < 16; ++t) acc[t] = v8f_zero();
    #pragma unroll
    for (int c = 0; c < 8; ++c){
      const v16bf a = gather_afrag(Bu, 258, 0, c << 5, lane);
      #pragma unroll
      for (int t = 0; t < 16; ++t)
        acc[t] = wmma_bf16(a, load_bfrag(wsw, FB_WM2 + (c << 4) + t, lane), acc[t]);
    }
    epilogue_store<16, false>(acc, mb2, mb2, mb2, A, 258, 0, lane);
  }
  __syncthreads();

  // model layer 3: 256 -> 5, VALU GEMV
  for (int o = lane; o < 80; o += 32){
    const int s = o / 5, a = o % 5;
    float accv = b3[a];
    for (int k = 0; k < 256; ++k) accv += bf2f(A[s * 258 + k]) * W3[k * 5 + a];
    out[(size_t)(sb + s) * 5 + a] = accv;
  }
}

extern "C" void kernel_launch(void* const* d_in, const int* in_sizes, int n_in,
                              void* d_out, int out_size, void* d_ws, size_t ws_size,
                              hipStream_t stream) {
  (void)in_sizes; (void)n_in; (void)out_size; (void)ws_size;
  const float* sur_data = (const float*)d_in[0];
  const int*   sur_n    = (const int*)  d_in[1];
  const float* ego_data = (const float*)d_in[2];
  const int*   ego_n    = (const int*)  d_in[3];
  const float* sur_mean = (const float*)d_in[4];
  const float* sur_std  = (const float*)d_in[5];
  const float* ego_mean = (const float*)d_in[6];
  const float* ego_std  = (const float*)d_in[7];
  const float* sW1  = (const float*)d_in[8];
  const float* sb1  = (const float*)d_in[9];
  const float* sg1  = (const float*)d_in[10];
  const float* sbb1 = (const float*)d_in[11];
  const float* sW2  = (const float*)d_in[12];
  const float* sb2  = (const float*)d_in[13];
  const float* sg2  = (const float*)d_in[14];
  const float* sbb2 = (const float*)d_in[15];
  const float* eW1  = (const float*)d_in[16];
  const float* eb1  = (const float*)d_in[17];
  const float* eg1  = (const float*)d_in[18];
  const float* ebb1 = (const float*)d_in[19];
  const float* eW2  = (const float*)d_in[20];
  const float* eb2  = (const float*)d_in[21];
  const float* eg2  = (const float*)d_in[22];
  const float* ebb2 = (const float*)d_in[23];
  const float* pW1  = (const float*)d_in[24];
  const float* pb1  = (const float*)d_in[25];
  const float* pg1  = (const float*)d_in[26];
  const float* pbb1 = (const float*)d_in[27];
  const float* pW2  = (const float*)d_in[28];
  const float* pb2  = (const float*)d_in[29];
  const float* pg2  = (const float*)d_in[30];
  const float* pbb2 = (const float*)d_in[31];
  const float* mW1  = (const float*)d_in[32];
  const float* mb1  = (const float*)d_in[33];
  const float* mW2  = (const float*)d_in[34];
  const float* mb2  = (const float*)d_in[35];
  const float* mW3  = (const float*)d_in[36];
  const float* mb3  = (const float*)d_in[37];
  float* out = (float*)d_out;

  // workspace layout: [0, 512KB) swizzled bf16 weights; then sur_feat f32 [8192,128]; ego_feat f32 [8192,64]
  unsigned short* wsw = (unsigned short*)d_ws;
  float* surF = (float*)((char*)d_ws + (1u << 19));
  float* egoF = (float*)((char*)d_ws + (1u << 19) + (size_t)8192 * 128 * 4);

  prep_kernel<<<FRAGS_TOTAL, 32, 0, stream>>>(sW1, sW2, pW1, pW2, mW1, mW2, wsw);
  sur_kernel<<<8192, 256, 0, stream>>>(sur_data, sur_n, sur_mean, sur_std,
                                       sb1, sg1, sbb1, sb2, sg2, sbb2, wsw, surF);
  ego_kernel<<<8192, 64, 0, stream>>>(ego_data, ego_n, ego_mean, ego_std,
                                      eW1, eb1, eg1, ebb1, eW2, eb2, eg2, ebb2, egoF);
  head_kernel<<<256, 64, 0, stream>>>(surF, egoF, wsw,
                                      pb1, pg1, pbb1, pb2, pg2, pbb2,
                                      mb1, mb2, mW3, mb3, out);
}
